// SocialAttention_23270132809876
// MI455X (gfx1250) — matrix operand order repaired and checked by reading, not compiled, not run on hardware
//
#include <hip/hip_runtime.h>
#include <hip/hip_bf16.h>
#include <math.h>

// ---------------- sizes ----------------
#define BB 2
#define AA 512
#define DD 512
#define HH 8
#define HD 64
#define D3 1536
#define RH 512
#define SCALE 0.125f   // HD^-0.5

typedef __attribute__((ext_vector_type(16))) __bf16 v16bf;
typedef __attribute__((ext_vector_type(8)))  float  v8f;

// ---------------- WMMA helpers ----------------
__device__ __forceinline__ v8f wmma_bf16(v16bf a, v16bf b, v8f c) {
  return __builtin_amdgcn_wmma_f32_16x16x32_bf16(false, a, false, b, (short)0, c, false, false);
}

// A-fragment (16x32 bf16, M = lane&15, K per ISA table), row-major source
__device__ __forceinline__ v16bf load_afrag(const __bf16* base, int ldk, int row, int k0, int lane) {
  const __bf16* p = base + (size_t)row * ldk + k0;
  const int o1 = (lane & 16) ? 8 : 0;
  const int o2 = (lane & 16) ? 24 : 16;
  v16bf a;
#pragma unroll
  for (int e = 0; e < 8; ++e) a[e] = p[o1 + e];
#pragma unroll
  for (int e = 0; e < 8; ++e) a[8 + e] = p[o2 + e];
  return a;
}

// B-fragment (32x16 bf16, N = lane&15). rowptr points at the N-row of a
// row-major K-contiguous (i.e. transposed-weight) matrix: K per ISA table (e<8: k0+(hi?8:0)+e, e>=8: k0+(hi?24:16)+e-8)
__device__ __forceinline__ v16bf load_bfragT(const __bf16* rowptr, int k0, int lane) {
  const __bf16* p = rowptr + k0;
  const int o1 = (lane & 16) ? 8 : 0;
  const int o2 = (lane & 16) ? 24 : 16;
  v16bf b;
#pragma unroll
  for (int e = 0; e < 8; ++e) b[e] = p[o1 + e];
#pragma unroll
  for (int e = 0; e < 8; ++e) b[8 + e] = p[o2 + e];
  return b;
}

// ---------------- elementwise conversions ----------------
__global__ void k_cvt_bf16(const float* __restrict__ in, __bf16* __restrict__ out, int n) {
  int t = blockIdx.x * blockDim.x + threadIdx.x;
  if (t < n) out[t] = (__bf16)in[t];
}

// W (K x N, f32) -> Wt (Npad x K, bf16), zero-padded rows for n >= N
__global__ void k_tconv(const float* __restrict__ W, __bf16* __restrict__ Wt, int K, int N, int Npad) {
  int t = blockIdx.x * blockDim.x + threadIdx.x;
  if (t >= K * Npad) return;
  int n = t / K, k = t - n * K;
  Wt[(size_t)n * K + k] = (n < N) ? (__bf16)W[(size_t)k * N + n] : (__bf16)0.0f;
}

// ---------------- generic WMMA GEMM: C(MxN) = A(MxK) @ Bt(NxK)^T ----------------
// one wave computes a 32x64 tile (2 A-frags x 4 B-frags -> 8 WMMA per K-chunk)
__global__ void k_gemm(const __bf16* __restrict__ A, const __bf16* __restrict__ Bt,
                       const float* __restrict__ bias, float* __restrict__ Cf,
                       __bf16* __restrict__ Cb, int M, int N, int K, int relu) {
  const int lane = threadIdx.x;
  const int m0 = blockIdx.x * 32;
  const int n0 = blockIdx.y * 64;
  const int arow0 = m0 + (lane & 15);
  v8f acc[2][4] = {{v8f{}, v8f{}, v8f{}, v8f{}}, {v8f{}, v8f{}, v8f{}, v8f{}}};
  for (int k0 = 0; k0 < K; k0 += 32) {
    if (k0 + 128 < K) {  // prefetch a couple of chunks ahead (global_prefetch_b8)
      __builtin_prefetch(A + (size_t)arow0 * K + k0 + 128, 0, 3);
      __builtin_prefetch(A + (size_t)(arow0 + 16) * K + k0 + 128, 0, 3);
    }
    v16bf af0 = load_afrag(A, K, arow0, k0, lane);
    v16bf af1 = load_afrag(A, K, arow0 + 16, k0, lane);
#pragma unroll
    for (int t = 0; t < 4; ++t) {
      const __bf16* brow = Bt + (size_t)(n0 + t * 16 + (lane & 15)) * K;
      v16bf bf = load_bfragT(brow, k0, lane);
      acc[0][t] = wmma_bf16(af0, bf, acc[0][t]);
      acc[1][t] = wmma_bf16(af1, bf, acc[1][t]);
    }
  }
#pragma unroll
  for (int half = 0; half < 2; ++half) {
#pragma unroll
    for (int t = 0; t < 4; ++t) {
      int n = n0 + t * 16 + (lane & 15);
      float bv = bias ? bias[n] : 0.0f;
#pragma unroll
      for (int r = 0; r < 8; ++r) {
        int mi = m0 + half * 16 + r + ((lane & 16) ? 8 : 0);
        float v = acc[half][t][r] + bv;
        if (relu) v = fmaxf(v, 0.0f);
        if (Cf) Cf[(size_t)mi * N + n] = v;
        if (Cb) Cb[(size_t)mi * N + n] = (__bf16)v;
      }
    }
  }
}

// ---------------- qkv split: (B*A,1536) f32 -> q,k (B,H,A,HD) bf16 ; v transposed (B,H,HD,A) ----------------
__global__ void k_qkv_split(const float* __restrict__ qkvf, __bf16* __restrict__ qb,
                            __bf16* __restrict__ kb, __bf16* __restrict__ vtb) {
  int t = blockIdx.x * blockDim.x + threadIdx.x;   // 0 .. B*A*D-1
  if (t >= BB * AA * DD) return;
  int col = t & (DD - 1);
  int row = t >> 9;                 // token index in [0, B*A)
  int b = row >> 9, a = row & (AA - 1);
  int h = col >> 6, hd = col & (HD - 1);
  const float* qr = qkvf + (size_t)row * D3;
  size_t qi = ((size_t)(b * HH + h) * AA + a) * HD + hd;
  qb[qi] = (__bf16)(qr[col] * SCALE);
  kb[qi] = (__bf16)qr[DD + col];
  vtb[((size_t)(b * HH + h) * HD + hd) * AA + a] = (__bf16)qr[2 * DD + col];
}

// ---------------- fused relation MLP -> bias = 16*sigmoid(.) , (B,H,A,A) f32 ----------------
// layer1 (K=4) on VALU directly in A-fragment layout; layer2 (K=512) on WMMA
__global__ void k_relbias(const float* __restrict__ relation, const float* __restrict__ w1,
                          const float* __restrict__ b1, const __bf16* __restrict__ w2t,
                          float* __restrict__ relbias) {
  __shared__ float w1s[4 * RH];
  __shared__ float b1s[RH];
  for (int t = threadIdx.x; t < 4 * RH; t += blockDim.x) w1s[t] = w1[t];
  for (int t = threadIdx.x; t < RH; t += blockDim.x) b1s[t] = b1[t];
  __syncthreads();

  const int bi = blockIdx.x;                  // b*A + i
  const int b = bi >> 9, i = bi & (AA - 1);
  const int wave = threadIdx.x >> 5, lane = threadIdx.x & 31;

  for (int jt = wave * 4; jt < wave * 4 + 4; ++jt) {
    const int j0 = jt * 16;
    const int j = j0 + (lane & 15);
    const float* rp = relation + (((size_t)(b * AA + i)) * AA + j) * 4;
    const float r0 = rp[0], r1 = rp[1], r2 = rp[2], r3 = rp[3];
    v8f acc = {};
    for (int kc = 0; kc < 16; ++kc) {
      const int k0 = kc * 32;
      v16bf af;
#pragma unroll
      for (int e = 0; e < 16; ++e) {
        int hid = k0 + e + ((e < 8) ? ((lane & 16) ? 8 : 0) : ((lane & 16) ? 16 : 8));
        float hv = fmaf(r0, w1s[hid],
                   fmaf(r1, w1s[RH + hid],
                   fmaf(r2, w1s[2 * RH + hid],
                   fmaf(r3, w1s[3 * RH + hid], b1s[hid]))));
        af[e] = (__bf16)fmaxf(hv, 0.0f);
      }
      v16bf bf = load_bfragT(w2t + (size_t)(lane & 15) * RH, k0, lane);
      acc = wmma_bf16(af, bf, acc);
    }
    const int h = lane & 15;
    if (h < HH) {
#pragma unroll
      for (int r = 0; r < 8; ++r) {
        int jj = j0 + r + ((lane & 16) ? 8 : 0);
        float v = acc[r];
        relbias[(((size_t)(b * HH + h) * AA) + i) * AA + jj] = 16.0f / (1.0f + expf(-v));
      }
    }
  }
}

// ---------------- attention: one wave per (b,h,16-row tile) ----------------
__global__ void k_attn(const __bf16* __restrict__ qb, const __bf16* __restrict__ kb,
                       const __bf16* __restrict__ vtb, const float* __restrict__ relbias,
                       const unsigned char* __restrict__ mask, float* __restrict__ attn_out) {
  __shared__ float sc[16][513];     // 513-dword row stride -> bank spread
  __shared__ float invs[16];
  const int lane = threadIdx.x;
  const int bid = blockIdx.x;
  const int it = bid & 31;          // i-tile
  const int bh = bid >> 5;          // b*H + h
  const int h = bh & (HH - 1), b = bh >> 3;
  const int i0 = it * 16;

  const __bf16* qbase = qb + (size_t)bh * AA * HD;
  const __bf16* kbase = kb + (size_t)bh * AA * HD;
  const __bf16* vbase = vtb + (size_t)bh * HD * AA;
  const float* rb = relbias + ((size_t)bh * AA + i0) * AA;

  // phase 1: scores S = (q*scale) @ k^T + relbias + mask
  const int arow = i0 + (lane & 15);
  v16bf qf0 = load_afrag(qbase, HD, arow, 0, lane);    // j-invariant: hoisted
  v16bf qf1 = load_afrag(qbase, HD, arow, 32, lane);
  for (int jt = 0; jt < 32; ++jt) {
    v8f acc = {};
    const __bf16* krow = kbase + (size_t)(jt * 16 + (lane & 15)) * HD;
    acc = wmma_bf16(qf0, load_bfragT(krow, 0, lane), acc);
    acc = wmma_bf16(qf1, load_bfragT(krow, 32, lane), acc);
    int j = jt * 16 + (lane & 15);
    float mb = mask[b * AA + j] ? -1e9f : 0.0f;
#pragma unroll
    for (int r = 0; r < 8; ++r) {
      int mi = r + ((lane & 16) ? 8 : 0);
      sc[mi][j] = acc[r] + rb[(size_t)mi * AA + j] + mb;
    }
  }
  __syncthreads();

  // phase 2: softmax over j (each lane owns half a row, combine via xor-16 shuffle)
  {
    const int row = lane & 15;
    const int base = (lane & 16) ? 256 : 0;
    float mx = -3.4e38f;
    for (int t = 0; t < 256; ++t) mx = fmaxf(mx, sc[row][base + t]);
    mx = fmaxf(mx, __shfl_xor(mx, 16, 32));
    float sum = 0.0f;
    for (int t = 0; t < 256; ++t) {
      float p = expf(sc[row][base + t] - mx);
      sc[row][base + t] = p;
      sum += p;
    }
    sum += __shfl_xor(sum, 16, 32);
    if (lane < 16) invs[row] = 1.0f / sum;
  }
  __syncthreads();
  const float is_ = invs[lane & 15];

  // phase 3: O = P @ V  (V pre-transposed -> contiguous B-fragments)
  float* orow = attn_out + ((size_t)(b * AA) + i0) * DD + h * HD;
  for (int nc = 0; nc < 4; ++nc) {
    v8f acc = {};
    for (int kc = 0; kc < 16; ++kc) {
      v16bf af;
#pragma unroll
      for (int e = 0; e < 16; ++e) {
        int jj = kc * 32 + e + ((e < 8) ? ((lane & 16) ? 8 : 0) : ((lane & 16) ? 16 : 8));
        af[e] = (__bf16)(sc[lane & 15][jj] * is_);
      }
      v16bf bf = load_bfragT(vbase + (size_t)(nc * 16 + (lane & 15)) * AA, kc * 32, lane);
      acc = wmma_bf16(af, bf, acc);
    }
    int n = nc * 16 + (lane & 15);
#pragma unroll
    for (int r = 0; r < 8; ++r) {
      int mi = r + ((lane & 16) ? 8 : 0);
      orow[(size_t)mi * DD + n] = acc[r];
    }
  }
}

// ---------------- residual + layernorm (one wave per row of 512) ----------------
__global__ void k_ln(const float* __restrict__ A, const float* __restrict__ R,
                     const float* __restrict__ g, const float* __restrict__ be,
                     float* __restrict__ outF, __bf16* __restrict__ outB) {
  const int wave = threadIdx.x >> 5, lane = threadIdx.x & 31;
  const int row = blockIdx.x * 8 + wave;
  const size_t base = (size_t)row * DD;
  float v[16], s = 0.0f, ss = 0.0f;
#pragma unroll
  for (int t = 0; t < 16; ++t) {
    int c = t * 32 + lane;
    float x = A[base + c] + R[base + c];
    v[t] = x; s += x; ss += x * x;
  }
#pragma unroll
  for (int off = 16; off >= 1; off >>= 1) {
    s += __shfl_xor(s, off, 32);
    ss += __shfl_xor(ss, off, 32);
  }
  float mean = s * (1.0f / DD);
  float var = ss * (1.0f / DD) - mean * mean;
  float rs = rsqrtf(var + 1e-5f);
#pragma unroll
  for (int t = 0; t < 16; ++t) {
    int c = t * 32 + lane;
    float o = (v[t] - mean) * rs * g[c] + be[c];
    if (outF) outF[base + c] = o;
    if (outB) outB[base + c] = (__bf16)o;
  }
}

// ---------------- launch ----------------
extern "C" void kernel_launch(void* const* d_in, const int* in_sizes, int n_in,
                              void* d_out, int out_size, void* d_ws, size_t ws_size,
                              hipStream_t stream) {
  const float* x       = (const float*)d_in[0];
  const unsigned char* mask = (const unsigned char*)d_in[1];
  const float* relation= (const float*)d_in[2];
  const float* qkv_w   = (const float*)d_in[3];
  const float* qkv_b   = (const float*)d_in[4];
  const float* rel_w1  = (const float*)d_in[5];
  const float* rel_b1  = (const float*)d_in[6];
  const float* rel_w2  = (const float*)d_in[7];
  const float* ff_w1   = (const float*)d_in[8];
  const float* ff_b1   = (const float*)d_in[9];
  const float* ff_w2   = (const float*)d_in[10];
  const float* ff_b2   = (const float*)d_in[11];
  const float* ln1_g   = (const float*)d_in[12];
  const float* ln1_b   = (const float*)d_in[13];
  const float* ln2_g   = (const float*)d_in[14];
  const float* ln2_b   = (const float*)d_in[15];
  float* outp = (float*)d_out;

  char* w = (char*)d_ws;
  __bf16* xb     = (__bf16*)w; w += (size_t)BB*AA*DD*2;
  __bf16* qkvwt  = (__bf16*)w; w += (size_t)D3*DD*2;
  __bf16* ffw1t  = (__bf16*)w; w += (size_t)D3*DD*2;
  __bf16* ffw2t  = (__bf16*)w; w += (size_t)DD*D3*2;
  __bf16* relw2t = (__bf16*)w; w += (size_t)16*RH*2;
  __bf16* qb     = (__bf16*)w; w += (size_t)BB*HH*AA*HD*2;
  __bf16* kb     = (__bf16*)w; w += (size_t)BB*HH*AA*HD*2;
  __bf16* vtb    = (__bf16*)w; w += (size_t)BB*HH*HD*AA*2;
  float*  qkvf   = (float*)w;  w += (size_t)BB*AA*D3*4;
  float*  relbias= (float*)w;  w += (size_t)BB*HH*AA*AA*4;
  float*  attn   = (float*)w;  w += (size_t)BB*AA*DD*4;
  float*  x1f    = (float*)w;  w += (size_t)BB*AA*DD*4;
  __bf16* x1b    = (__bf16*)w; w += (size_t)BB*AA*DD*2;
  __bf16* hb     = (__bf16*)w; w += (size_t)BB*AA*D3*2;
  float*  ffout  = (float*)w;  w += (size_t)BB*AA*DD*4;

  const int ne = BB*AA*DD;  // 524288

  // conversions / weight transposes
  k_cvt_bf16<<<(ne+255)/256, 256, 0, stream>>>(x, xb, ne);
  k_tconv<<<(DD*D3+255)/256, 256, 0, stream>>>(qkv_w, qkvwt, DD, D3, D3);
  k_tconv<<<(DD*D3+255)/256, 256, 0, stream>>>(ff_w1, ffw1t, DD, D3, D3);
  k_tconv<<<(D3*DD+255)/256, 256, 0, stream>>>(ff_w2, ffw2t, D3, DD, DD);
  k_tconv<<<(RH*16+255)/256, 256, 0, stream>>>(rel_w2, relw2t, RH, HH, 16);

  // qkv projection + split
  k_gemm<<<dim3((BB*AA)/32, D3/64), 32, 0, stream>>>(xb, qkvwt, qkv_b, qkvf, nullptr,
                                                     BB*AA, D3, DD, 0);
  k_qkv_split<<<(ne+255)/256, 256, 0, stream>>>(qkvf, qb, kb, vtb);

  // relation MLP bias
  k_relbias<<<BB*AA, 256, 0, stream>>>(relation, rel_w1, rel_b1, relw2t, relbias);

  // attention
  k_attn<<<BB*HH*(AA/16), 32, 0, stream>>>(qb, kb, vtb, relbias, mask, attn);

  // LN1 (attn + residual x)
  k_ln<<<(BB*AA)/8, 256, 0, stream>>>(attn, x, ln1_g, ln1_b, x1f, x1b);

  // FFN
  k_gemm<<<dim3((BB*AA)/32, D3/64), 32, 0, stream>>>(x1b, ffw1t, ff_b1, nullptr, hb,
                                                     BB*AA, D3, DD, 1);
  k_gemm<<<dim3((BB*AA)/32, DD/64), 32, 0, stream>>>(hb, ffw2t, ff_b2, ffout, nullptr,
                                                     BB*AA, DD, D3, 0);

  // LN2 -> output
  k_ln<<<(BB*AA)/8, 256, 0, stream>>>(x1f, ffout, ln2_g, ln2_b, outp, nullptr);
}
